// EventGRUBitLevel_83090437308605
// MI455X (gfx1250) — compile-verified, matrix operand order checked by reading, not run on hardware
//
#include <hip/hip_runtime.h>

// ---------------------------------------------------------------------------
// EventGRUBitLevel for MI455X (gfx1250, wave32)
// B=4096, IN_DIM=256, T=64, E=32, H=256, 3H=768
//
//  - bf16 WMMA (v_wmma_f32_16x16x32_bf16) for all GEMMs, f32 accumulate.
//  - 128 WGs x 256 threads (8 waves). WG owns 32 batch rows; wave = (mrow
//    half) x (jt quarter of the 768 gate columns) -> 1024 waves total.
//  - W_hh packed bf16 kt-major; each K=32 slice (48KB) staged into double-
//    buffered LDS by the Tensor Data Mover (tensor_load_to_lds +
//    s_wait_tensorcnt), consumed by all waves -> 8x less L2 traffic.
//  - h ping-ponged in LDS as bf16; per-row decoder dot reduced via
//    shfl_xor + ds_add_f32; E=32 exclusive cumsum = wave32 prefix scan.
// ---------------------------------------------------------------------------

typedef __attribute__((ext_vector_type(16))) __bf16 v16bf;
typedef __attribute__((ext_vector_type(8)))  float  v8f;
typedef __attribute__((ext_vector_type(4)))  unsigned int u32x4;
typedef __attribute__((ext_vector_type(8)))  int  i32x8;
typedef __attribute__((ext_vector_type(4)))  int  i32x4;

#ifndef __has_builtin
#define __has_builtin(x) 0
#endif
#if __has_builtin(__builtin_amdgcn_tensor_load_to_lds) && \
    __has_builtin(__builtin_amdgcn_s_wait_tensorcnt)
#define HAVE_TDM 1
#else
#define HAVE_TDM 0
#endif

__device__ __forceinline__ unsigned short f2bf(float x) {
  unsigned int u = __builtin_bit_cast(unsigned int, x);
  u = (u + 0x7FFFu + ((u >> 16) & 1u)) >> 16;
  return (unsigned short)u;
}
__device__ __forceinline__ float bf2f(unsigned short h) {
  return __builtin_bit_cast(float, ((unsigned int)h) << 16);
}

struct U32x8 { uint4 lo, hi; };

// A-fragment load (row-major bf16, chunks at +0 and +16 halves)
__device__ __forceinline__ v16bf load_frag_split(const unsigned short* p) {
  U32x8 u;
  u.lo = *(const uint4*)(p);
  u.hi = *(const uint4*)(p + 16);
  return __builtin_bit_cast(v16bf, u);
}
// Packed B-fragment load: 16 contiguous halves per lane
__device__ __forceinline__ v16bf load_frag_packed(const unsigned short* p) {
  U32x8 u;
  u.lo = *(const uint4*)(p);
  u.hi = *(const uint4*)(p + 8);
  return __builtin_bit_cast(v16bf, u);
}

__device__ __forceinline__ v8f bfwmma(v16bf a, v16bf b, v8f c) {
  return __builtin_amdgcn_wmma_f32_16x16x32_bf16(false, a, false, b,
                                                 (short)0, c, false, false);
}

__device__ __forceinline__ float sigmoidf_fast(float x) {
  return 1.0f / (1.0f + __expf(-x));
}

#if HAVE_TDM
// TDM: stage one contiguous 48KB chunk (6144 x 8B) global -> LDS.
__device__ __forceinline__ void tdm_stage_48k(unsigned lds_byte,
                                              const void* gsrc) {
  unsigned long long ga = (unsigned long long)(size_t)gsrc;
  u32x4 g0;
  g0.x = 1u;                                   // count = 1 (valid descriptor)
  g0.y = lds_byte;                             // lds_addr
  g0.z = (unsigned)(ga & 0xFFFFFFFFu);         // global_addr[31:0]
  g0.w = (unsigned)((ga >> 32) & 0x01FFFFFFu)  // global_addr[56:32]
         | (2u << 30);                         // type = 2 (image)
  i32x8 g1;
  g1[0] = (3 << 16);                           // data_size = 8B, mask = 0
  g1[1] = (int)(6144u << 16);                  // tensor_dim0[15:0] @ w1[31:16]
  g1[2] = (int)(1u << 16);                     // dim0 hi=0; tensor_dim1 = 1
  g1[3] = (int)(6144u << 16);                  // dim1 hi=0; tile_dim0 = 6144
  g1[4] = 0;                                   // tile_dim1 = tile_dim2 = 0
  g1[5] = 6144;                                // tensor_dim0_stride lo
  g1[6] = 0;
  g1[7] = 0;
  i32x4 z4 = {0, 0, 0, 0};
#if __clang_major__ >= 23
  i32x8 z8 = {0, 0, 0, 0, 0, 0, 0, 0};
  __builtin_amdgcn_tensor_load_to_lds(g0, g1, z4, z4, z8, 0);
#else
  __builtin_amdgcn_tensor_load_to_lds(g0, g1, z4, z4, 0);
#endif
}
#define TDM_WAIT(n) __builtin_amdgcn_s_wait_tensorcnt((short)(n))
#endif

// ---------------------------------------------------------------------------
// Pack W (rows = N units, cols = K) into bf16 WMMA B-fragments.
// Fragment = 512 halves; lane l holds 16 halves:
//   n = l & 15,  k = (i&7) + ((l>>4)<<3) + ((i>>3)<<4)
// so B[k][n] = W[nt*16 + n][kt*32 + k].
// ktmajor != 0 -> fragment order (kt, nt) so a full K-slice is contiguous.
// ---------------------------------------------------------------------------
__global__ void pack_weights_kernel(const float* __restrict__ W,
                                    unsigned short* __restrict__ dst,
                                    int ntiles, int ktiles, int K,
                                    int ktmajor) {
  int total = ntiles * ktiles * 512;
  for (int idx = blockIdx.x * blockDim.x + threadIdx.x; idx < total;
       idx += gridDim.x * blockDim.x) {
    int frag = idx >> 9;
    int e    = idx & 511;
    int lane = e >> 4;
    int i    = e & 15;
    int kt, nt;
    if (ktmajor) { kt = frag / ntiles; nt = frag % ntiles; }
    else         { kt = frag % ktiles; nt = frag / ktiles; }
    int n = lane & 15;
    int k = (i & 7) + ((lane >> 4) << 3) + ((i >> 3) << 4);
    dst[idx] = f2bf(W[(nt * 16 + n) * K + (kt * 32 + k)]);
  }
}

// ---------------------------------------------------------------------------
// Main fused kernel: 128 blocks x 256 threads, 32 batch rows per block.
// ---------------------------------------------------------------------------
__global__ __launch_bounds__(256, 1)
void event_gru_fused_kernel(const float* __restrict__ x,
                            const int* __restrict__ targets,
                            const float* __restrict__ b_e1,
                            const float* __restrict__ b_e2,
                            const float* __restrict__ b_ih,
                            const float* __restrict__ b_hh,
                            const float* __restrict__ W_dec,
                            const float* __restrict__ b_dec,
                            const unsigned short* __restrict__ pE1,
                            const unsigned short* __restrict__ pE2,
                            const unsigned short* __restrict__ pIH,
                            const unsigned short* __restrict__ pHH,
                            float* __restrict__ out) {
  extern __shared__ char smemraw[];
  unsigned short* bufA  = (unsigned short*)(smemraw);            // 32x256
  unsigned short* bufB  = (unsigned short*)(smemraw + 16384);    // 32x256
  unsigned short* pbufA = (unsigned short*)(smemraw + 32768);    // 32x32
  unsigned short* pbufB = (unsigned short*)(smemraw + 34816);    // 32x32
  unsigned short* wst0  = (unsigned short*)(smemraw + 36864);    // 48KB
  unsigned short* wst1  = (unsigned short*)(smemraw + 86016);    // 48KB
  float*          baseL = (float*)(smemraw + 135168);            // 32 floats

  const int tid      = threadIdx.x;
  const int lane     = tid & 31;
  const int wave     = tid >> 5;
  const int mrow     = wave >> 2;                 // 0|1 : which 16-row tile
  const int q        = wave & 3;                  // jt quarter (4 jt each)
  const int mrowBase = mrow * 16;
  const int blockRow = blockIdx.x * 32;           // global batch row base
  const int halfSel  = lane >> 4;
  const int nl       = lane & 15;
  const int koff     = halfSel * 8;
  const int rowA     = mrowBase + nl;             // LDS row this lane feeds A

  // ---- stage x tile into bufA as bf16 ----
  for (int i = tid; i < 32 * 256; i += 256)
    bufA[i] = f2bf(x[(blockRow + (i >> 8)) * 256 + (i & 255)]);
  __syncthreads();

  v16bf afr[8];

  // ---- encoder layer 1: relu(x @ W_e1^T + b_e1) : bufA -> bufB ----
#pragma unroll
  for (int kt = 0; kt < 8; ++kt)
    afr[kt] = load_frag_split(&bufA[rowA * 256 + kt * 32 + koff]);
#pragma unroll
  for (int jq = 0; jq < 4; ++jq) {
    int jt = q * 4 + jq;
    v8f acc = {};
#pragma unroll
    for (int kt = 0; kt < 8; ++kt)
      acc = bfwmma(afr[kt],
                   load_frag_packed(&pE1[(jt * 8 + kt) * 512 + lane * 16]), acc);
    int   j    = jt * 16 + nl;
    float bias = b_e1[j];
#pragma unroll
    for (int v = 0; v < 8; ++v) {
      float val = acc[v] + bias;
      val = val > 0.f ? val : 0.f;
      bufB[(mrowBase + v + halfSel * 8) * 256 + j] = f2bf(val);
    }
  }
  __syncthreads();

  // ---- encoder layer 2: relu(h1 @ W_e2^T + b_e2) : bufB -> bufA ----
#pragma unroll
  for (int kt = 0; kt < 8; ++kt)
    afr[kt] = load_frag_split(&bufB[rowA * 256 + kt * 32 + koff]);
#pragma unroll
  for (int jq = 0; jq < 4; ++jq) {
    int jt = q * 4 + jq;
    v8f acc = {};
#pragma unroll
    for (int kt = 0; kt < 8; ++kt)
      acc = bfwmma(afr[kt],
                   load_frag_packed(&pE2[(jt * 8 + kt) * 512 + lane * 16]), acc);
    int   j    = jt * 16 + nl;
    float bias = b_e2[j];
#pragma unroll
    for (int v = 0; v < 8; ++v) {
      float val = acc[v] + bias;
      val = val > 0.f ? val : 0.f;
      bufA[(mrowBase + v + halfSel * 8) * 256 + j] = f2bf(val);
    }
  }

  // ---- prev0 = 0 ----
  for (int i = tid; i < 32 * 32; i += 256) pbufA[i] = 0;
  __syncthreads();

  const float wb_lane = W_dec[256 + lane];        // w_b[e], e = lane
  const float b0      = b_dec[0];

  unsigned short* cur = bufA;
  unsigned short* nxt = bufB;
  unsigned short* pc  = pbufA;
  unsigned short* pn  = pbufB;
  const unsigned lds_w0 = (unsigned)(size_t)(void*)wst0;
  const unsigned lds_w1 = (unsigned)(size_t)(void*)wst1;

  for (int t = 0; t < 64; ++t) {
    if (wave == 7) baseL[lane] = 0.f;             // zero decoder accumulators

    // prev-logits A fragment + gi WMMAs (K = 32, B streamed from L2)
    v16bf pfr = load_frag_split(&pc[rowA * 32 + koff]);
    v8f accR[4], accZ[4], accGN[4], accIN[4];
#pragma unroll
    for (int jq = 0; jq < 4; ++jq) {
      int jt = q * 4 + jq;
      v8f zz = {};
      accGN[jq] = zz;
      accR[jq]  = bfwmma(pfr, load_frag_packed(&pIH[(jt)      * 512 + lane * 16]), zz);
      accZ[jq]  = bfwmma(pfr, load_frag_packed(&pIH[(16 + jt) * 512 + lane * 16]), zz);
      accIN[jq] = bfwmma(pfr, load_frag_packed(&pIH[(32 + jt) * 512 + lane * 16]), zz);
    }

    // gh GEMM: kt-outer, W_hh K-slices TDM-staged to LDS, double buffered
    for (int kt = 0; kt < 8; ++kt) {
#if HAVE_TDM
      if (wave == 0) {
        if (kt == 0)
          tdm_stage_48k(lds_w0, (const void*)(pHH + 0));
        if (kt + 1 < 8)
          tdm_stage_48k(((kt + 1) & 1) ? lds_w1 : lds_w0,
                        (const void*)(pHH + (size_t)(kt + 1) * 24576));
        if (kt + 1 < 8) TDM_WAIT(1); else TDM_WAIT(0);
      }
      __syncthreads();
#else
      __syncthreads();
      {
        unsigned short* db = (kt & 1) ? wst1 : wst0;
        const uint4* src = (const uint4*)(pHH + (size_t)kt * 24576);
        uint4* dsp = (uint4*)db;
        for (int i = tid; i < 3072; i += 256) dsp[i] = src[i];
      }
      __syncthreads();
#endif
      const unsigned short* wb = (kt & 1) ? wst1 : wst0;
      v16bf af = load_frag_split(&cur[rowA * 256 + kt * 32 + koff]);
#pragma unroll
      for (int jq = 0; jq < 4; ++jq) {
        int jt = q * 4 + jq;
        accR[jq]  = bfwmma(af, load_frag_packed(&wb[(jt)      * 512 + lane * 16]), accR[jq]);
        accZ[jq]  = bfwmma(af, load_frag_packed(&wb[(16 + jt) * 512 + lane * 16]), accZ[jq]);
        accGN[jq] = bfwmma(af, load_frag_packed(&wb[(32 + jt) * 512 + lane * 16]), accGN[jq]);
      }
      __syncthreads();                            // buffer reuse fence
    }

    // gate nonlinearity + h update + decoder partial dot
    v8f base_acc = {};
#pragma unroll
    for (int jq = 0; jq < 4; ++jq) {
      int         jt  = q * 4 + jq;
      const int   j   = jt * 16 + nl;
      const float br  = b_ih[j]       + b_hh[j];
      const float bz  = b_ih[256 + j] + b_hh[256 + j];
      const float bni = b_ih[512 + j];
      const float bnh = b_hh[512 + j];
      const float wh  = W_dec[j];
#pragma unroll
      for (int v = 0; v < 8; ++v) {
        float r = sigmoidf_fast(accR[jq][v] + br);
        float z = sigmoidf_fast(accZ[jq][v] + bz);
        float n = tanhf(accIN[jq][v] + bni + r * (accGN[jq][v] + bnh));
        int   row  = mrowBase + v + halfSel * 8;
        float hold = bf2f(cur[row * 256 + j]);
        float hnew = (1.f - z) * n + z * hold;
        nxt[row * 256 + j] = f2bf(hnew);
        base_acc[v] += hnew * wh;
      }
    }
    // reduce over the 16 columns held by this half-wave, then ds_add_f32
#pragma unroll
    for (int v = 0; v < 8; ++v) {
      float s = base_acc[v];
      s += __shfl_xor(s, 1);
      s += __shfl_xor(s, 2);
      s += __shfl_xor(s, 4);
      s += __shfl_xor(s, 8);
      base_acc[v] = s;
    }
    if (nl == 0) {
#pragma unroll
      for (int v = 0; v < 8; ++v)
        atomicAdd(&baseL[mrowBase + halfSel * 8 + v], base_acc[v]);
    }
    __syncthreads();                              // base complete

    // logits: 4 rows per wave; E=32 exclusive cumsum = wave32 prefix scan
#pragma unroll
    for (int rr = 0; rr < 4; ++rr) {
      int   lrow = wave * 4 + rr;
      float bval = baseL[lrow] + b0;
      int   grow = blockRow + lrow;
      int   tb   = targets[grow * (64 * 32) + t * 32 + lane];
      float w    = tb ? wb_lane : 0.f;
      float s    = w;
#pragma unroll
      for (int off = 1; off < 32; off <<= 1) {
        float tmp = __shfl_up(s, off);
        if (lane >= off) s += tmp;
      }
      float logit = bval + (s - w);
      out[grow * (64 * 32) + t * 32 + lane] = logit;
      pn[lrow * 32 + lane] = f2bf(logit);
    }
    __syncthreads();                              // step complete

    unsigned short* tsw = cur; cur = nxt; nxt = tsw;
    tsw = pc; pc = pn; pn = tsw;
  }
}

// ---------------------------------------------------------------------------
extern "C" void kernel_launch(void* const* d_in, const int* in_sizes, int n_in,
                              void* d_out, int out_size, void* d_ws, size_t ws_size,
                              hipStream_t stream) {
  (void)in_sizes; (void)n_in; (void)out_size; (void)ws_size;
  const float* x      = (const float*)d_in[0];
  const int*   tgt    = (const int*)d_in[1];
  const float* W_e1   = (const float*)d_in[2];
  const float* b_e1   = (const float*)d_in[3];
  const float* W_e2   = (const float*)d_in[4];
  const float* b_e2   = (const float*)d_in[5];
  const float* W_ih   = (const float*)d_in[6];
  const float* b_ih   = (const float*)d_in[7];
  const float* W_hh   = (const float*)d_in[8];
  const float* b_hh   = (const float*)d_in[9];
  const float* W_dec  = (const float*)d_in[10];
  const float* b_dec  = (const float*)d_in[11];

  unsigned short* ws  = (unsigned short*)d_ws;
  unsigned short* pE1 = ws;                         // 16*8*512 = 65536 halves
  unsigned short* pE2 = pE1 + 16 * 8 * 512;         // 65536
  unsigned short* pIH = pE2 + 16 * 8 * 512;         // 48*512   = 24576
  unsigned short* pHH = pIH + 48 * 512;             // 48*8*512 = 196608 (kt-major)

  pack_weights_kernel<<<64,  256, 0, stream>>>(W_e1, pE1, 16, 8, 256, 0);
  pack_weights_kernel<<<64,  256, 0, stream>>>(W_e2, pE2, 16, 8, 256, 0);
  pack_weights_kernel<<<48,  256, 0, stream>>>(W_ih, pIH, 48, 1, 32, 0);
  pack_weights_kernel<<<192, 256, 0, stream>>>(W_hh, pHH, 48, 8, 256, 1);

  const size_t shbytes = 135168 + 32 * sizeof(float);   // ~132 KB
  hipFuncSetAttribute((const void*)event_gru_fused_kernel,
                      hipFuncAttributeMaxDynamicSharedMemorySize, (int)shbytes);

  event_gru_fused_kernel<<<128, 256, shbytes, stream>>>(
      x, tgt, b_e1, b_e2, b_ih, b_hh, W_dec, b_dec,
      pE1, pE2, pIH, pHH, (float*)d_out);
}
